// AtariNetwork_17119739642083
// MI455X (gfx1250) — matrix operand order repaired and checked by reading, not compiled, not run on hardware
//
#include <hip/hip_runtime.h>

// ---------------------------------------------------------------------------
// Types / helpers for CDNA5 (gfx1250) bf16 WMMA
// ---------------------------------------------------------------------------
typedef __attribute__((ext_vector_type(16))) __bf16 v16bf;
typedef __attribute__((ext_vector_type(8)))  float  v8f;
typedef __attribute__((ext_vector_type(4))) unsigned tdm_g0_t;
typedef __attribute__((ext_vector_type(8))) int      tdm_g1_t;
typedef __attribute__((ext_vector_type(4))) int      tdm_gx_t;
typedef __attribute__((ext_vector_type(8))) int      tdm_gy_t;

#define DEV __device__ __forceinline__

DEV v8f wmma_bf16(v16bf a, v16bf b, v8f c) {
  // D = A(16x32 bf16) * B(32x16 bf16) + C(16x16 f32)
  return __builtin_amdgcn_wmma_f32_16x16x32_bf16(false, a, false, b,
                                                 (short)0, c, false, false);
}

// Hardware packed f32->bf16 (RNE): dst = {bf16(lo), bf16(hi)}
DEV unsigned cvtpk(float lo, float hi) {
  unsigned r;
  asm("v_cvt_pk_bf16_f32 %0, %1, %2" : "=v"(r) : "v"(lo), "v"(hi));
  return r;
}
DEV void st_bf(__bf16* p, float z) {
  unsigned r = cvtpk(z, z);
  union { unsigned u; unsigned short s[2]; } c; c.u = r;
  union { unsigned short s; __bf16 b; } o; o.s = c.s[0];
  *p = o.b;
}

// Fragment layout (CDNA5 ISA 7.12.2), 16-bit A 16x32:
//   lane<16: row=lane, K runs {0..7,16..23}; lane>=16: row=lane-16, {8..15,24..31}
//   -> two contiguous 8-element (16B) runs at khalf and khalf+16.
// B 32x16: col=lane&15, K runs as A: 8 at kb=(lane>>4)*8, 8 at kb+16 -> two 16B runs.
union FragBits { uint4 q[2]; v16bf b; unsigned u[8]; };

DEV v16bf ld_frag2(const __bf16* p0, const __bf16* p1) {
  FragBits f;
  f.q[0] = *(const uint4*)p0;
  f.q[1] = *(const uint4*)p1;
  return f.b;
}

// ---------------------------------------------------------------------------
// Tensor Data Mover: 2D tile (bf16 elements) global -> LDS.
// D# per CDNA5 ISA. Groups 2/3 zero (<=2D tile).
// Issue from ONE wave per block (TDM ignores EXEC; issues once per wave).
// clang-23 builtin arity: (u32x4, i32x8, i32x4, i32x4, i32x8, i32 cpol).
// ---------------------------------------------------------------------------
DEV void tdm_load_2d_bf16(unsigned lds_addr, const void* gaddr,
                          unsigned tile_d0, unsigned tile_d1,
                          unsigned tensor_d0, unsigned tensor_d1,
                          unsigned d0_stride) {
  unsigned long long ga = (unsigned long long)gaddr;
  tdm_g0_t g0;
  g0[0] = 1u;                                      // count=1, user descriptor
  g0[1] = lds_addr;                                // lds_addr [63:32]
  g0[2] = (unsigned)(ga & 0xffffffffu);            // global_addr [95:64]
  g0[3] = (unsigned)((ga >> 32) & 0x01ffffffu)     // global_addr [120:96]
          | 0x80000000u;                           // type=2 ("image") [127:126]
  tdm_g1_t g1;
  g1[0] = (int)(1u << 16);                         // data_size=1 -> 2 bytes
  g1[1] = (int)((tensor_d0 & 0xffffu) << 16);      // tensor_dim0 [79:48]
  g1[2] = (int)(((tensor_d0 >> 16) & 0xffffu) |
                ((tensor_d1 & 0xffffu) << 16));    // ..., tensor_dim1 [111:80]
  g1[3] = (int)(((tensor_d1 >> 16) & 0xffffu) |
                ((tile_d0 & 0xffffu) << 16));      // ..., tile_dim0 [127:112]
  g1[4] = (int)(tile_d1 & 0xffffu);                // tile_dim1; tile_dim2=0
  g1[5] = (int)d0_stride;                          // tensor_dim0_stride lo32
  g1[6] = 0;                                       // stride hi / dim1_stride
  g1[7] = 0;
  tdm_gx_t z4 = {0, 0, 0, 0};
  tdm_gy_t z8 = {0, 0, 0, 0, 0, 0, 0, 0};
  __builtin_amdgcn_tensor_load_to_lds(g0, g1, z4, z4, z8, 0);
}

DEV unsigned lds_addr_of(const void* p) {          // generic ptr low 32 = LDS off
  return (unsigned)(unsigned long long)p;
}

// ---------------------------------------------------------------------------
// 0) fp32 -> bf16 convert with transpose:  out[n*K+k] = in[k*N+n]
// ---------------------------------------------------------------------------
__global__ void k_cvtT(const float* __restrict__ in, __bf16* __restrict__ out,
                       int K, int N) {
  int total = K * N;
  for (int i = blockIdx.x * blockDim.x + threadIdx.x; i < total;
       i += gridDim.x * blockDim.x) {
    int k = i / N, n = i - k * N;
    st_bf(out + n * K + k, in[i]);
  }
}

// ---------------------------------------------------------------------------
// 1) conv1: 84x84x4 -> 20x20x32, 8x8 stride 4.  K=256, N=32, M=819200.
//    2 M-tiles x 2 N-frags per wave (4 wmma / k-step). Weights via TDM.
// ---------------------------------------------------------------------------
DEV v16bf c1_afrag(const float* src, int k0) {     // k0 = ks*32 + khalf
  FragBits f;
#pragma unroll
  for (int g = 0; g < 2; ++g) {
    int k = k0 + g * 16;                           // k%32 in {0,8,16,24} -> kc=0
    int kh = k >> 5, kw = (k & 31) >> 2;
    const float* p = src + (kh * 84 + kw) * 4;     // 8 consecutive floats
    float4 t0 = *(const float4*)p;
    float4 t1 = *(const float4*)(p + 4);
    f.u[g * 4 + 0] = cvtpk(t0.x, t0.y);
    f.u[g * 4 + 1] = cvtpk(t0.z, t0.w);
    f.u[g * 4 + 2] = cvtpk(t1.x, t1.y);
    f.u[g * 4 + 3] = cvtpk(t1.z, t1.w);
  }
  return f.b;
}

__global__ __launch_bounds__(256) void k_conv1(const float* __restrict__ in,
                                               const __bf16* __restrict__ wT, // [32][256]
                                               const float* __restrict__ bias,
                                               __bf16* __restrict__ out) {
  __shared__ __bf16 wt[32 * 256];                  // [n][k] bf16, 16 KB
  if (threadIdx.x < 32) {                          // wave 0 issues the TDM
    tdm_load_2d_bf16(lds_addr_of(wt), wT, 256, 32, 256, 32, 256);
    __builtin_amdgcn_s_wait_tensorcnt((short)0);
  }
  __syncthreads();

  const int lane = threadIdx.x & 31;
  const int pair = blockIdx.x * 8 + (threadIdx.x >> 5);
  const int mt0 = pair * 2, mt1 = mt0 + 1;
  const int row = lane & 15, khalf = (lane >> 4) * 8;
  const int col = lane & 15, kb = (lane >> 4) * 8;

  int mA = mt0 * 16 + row;
  int mB = mt1 * 16 + row;
  int btA = mA / 400, pA = mA - btA * 400, oyA = pA / 20, oxA = pA - oyA * 20;
  int btB = mB / 400, pB = mB - btB * 400, oyB = pB / 20, oxB = pB - oyB * 20;
  const float* srcA = in + (((btA * 84 + oyA * 4) * 84) + oxA * 4) * 4;
  const float* srcB = in + (((btB * 84 + oyB * 4) * 84) + oxB * 4) * 4;
  const __bf16* br0 = wt + col * 256;
  const __bf16* br1 = wt + (16 + col) * 256;

  v8f a00 = {}, a01 = {}, a10 = {}, a11 = {};
  for (int ks = 0; ks < 8; ++ks) {
    v16bf fa0 = c1_afrag(srcA, ks * 32 + khalf);
    v16bf fa1 = c1_afrag(srcB, ks * 32 + khalf);
    v16bf fb0 = ld_frag2(br0 + ks * 32 + kb, br0 + ks * 32 + kb + 16);
    v16bf fb1 = ld_frag2(br1 + ks * 32 + kb, br1 + ks * 32 + kb + 16);
    a00 = wmma_bf16(fa0, fb0, a00);
    a01 = wmma_bf16(fa0, fb1, a01);
    a10 = wmma_bf16(fa1, fb0, a10);
    a11 = wmma_bf16(fa1, fb1, a11);
  }
  const int rb = (lane >> 4) * 8;
  float bz0 = bias[col], bz1 = bias[16 + col];
#pragma unroll
  for (int v = 0; v < 8; ++v) {
    int r0 = mt0 * 16 + rb + v, r1 = mt1 * 16 + rb + v;
    float z;
    z = a00[v] + bz0; st_bf(out + r0 * 32 + col,      z > 0.f ? z : 0.f);
    z = a01[v] + bz1; st_bf(out + r0 * 32 + 16 + col, z > 0.f ? z : 0.f);
    z = a10[v] + bz0; st_bf(out + r1 * 32 + col,      z > 0.f ? z : 0.f);
    z = a11[v] + bz1; st_bf(out + r1 * 32 + 16 + col, z > 0.f ? z : 0.f);
  }
}

// ---------------------------------------------------------------------------
// 2) conv2: 20x20x32 -> 9x9x64, 4x4 stride 2.  K=512, N=64, M=165888.
//    2 M-tiles x 4 N-frags per wave (8 wmma / k-step). Weights via TDM.
// ---------------------------------------------------------------------------
DEV const __bf16* c2addr(const __bf16* src, int k) {
  int kc = k & 31, kw = (k >> 5) & 3, kh = k >> 7;
  return src + (kh * 20 + kw) * 32 + kc;           // 16B aligned (kc mult of 8)
}

__global__ __launch_bounds__(256) void k_conv2(const __bf16* __restrict__ in,
                                               const __bf16* __restrict__ wT, // [64][512]
                                               const float* __restrict__ bias,
                                               __bf16* __restrict__ out) {
  __shared__ __bf16 wt[64 * 512];                  // [n][k], 64 KB
  if (threadIdx.x < 32) {
    tdm_load_2d_bf16(lds_addr_of(wt), wT, 512, 64, 512, 64, 512);
    __builtin_amdgcn_s_wait_tensorcnt((short)0);
  }
  __syncthreads();

  const int lane = threadIdx.x & 31;
  const int pair = blockIdx.x * 8 + (threadIdx.x >> 5);
  const int mt0 = pair * 2, mt1 = mt0 + 1;
  const int row = lane & 15, khalf = (lane >> 4) * 8;
  const int col = lane & 15, kb = (lane >> 4) * 8;

  int mA = mt0 * 16 + row, mB = mt1 * 16 + row;
  int btA = mA / 81, pA = mA - btA * 81, oyA = pA / 9, oxA = pA - oyA * 9;
  int btB = mB / 81, pB = mB - btB * 81, oyB = pB / 9, oxB = pB - oyB * 9;
  const __bf16* srcA = in + (((btA * 20 + oyA * 2) * 20) + oxA * 2) * 32;
  const __bf16* srcB = in + (((btB * 20 + oyB * 2) * 20) + oxB * 2) * 32;

  v8f acc[2][4] = {{{}, {}, {}, {}}, {{}, {}, {}, {}}};
  for (int ks = 0; ks < 16; ++ks) {
    int k0 = ks * 32 + khalf;
    v16bf fa0 = ld_frag2(c2addr(srcA, k0), c2addr(srcA, k0 + 16));
    v16bf fa1 = ld_frag2(c2addr(srcB, k0), c2addr(srcB, k0 + 16));
#pragma unroll
    for (int f = 0; f < 4; ++f) {
      const __bf16* brow = wt + (f * 16 + col) * 512 + ks * 32 + kb;
      v16bf fb = ld_frag2(brow, brow + 16);
      acc[0][f] = wmma_bf16(fa0, fb, acc[0][f]);
      acc[1][f] = wmma_bf16(fa1, fb, acc[1][f]);
    }
  }
  const int rb = (lane >> 4) * 8;
#pragma unroll
  for (int f = 0; f < 4; ++f) {
    float bz = bias[f * 16 + col];
#pragma unroll
    for (int v = 0; v < 8; ++v) {
      float z0 = acc[0][f][v] + bz, z1 = acc[1][f][v] + bz;
      st_bf(out + (mt0 * 16 + rb + v) * 64 + f * 16 + col, z0 > 0.f ? z0 : 0.f);
      st_bf(out + (mt1 * 16 + rb + v) * 64 + f * 16 + col, z1 > 0.f ? z1 : 0.f);
    }
  }
}

// ---------------------------------------------------------------------------
// 3) conv3: 9x9x64 -> 7x7x64, 3x3 stride 1.  K=576, N=64, M=100352.
//    Weights [64][576]; two 288-deep halves fetched as real 2D TDM tiles.
// ---------------------------------------------------------------------------
DEV const __bf16* c3addr(const __bf16* src, int k) {
  int kc = k & 63, t = k >> 6, kw = t % 3, kh = t / 3;
  return src + (kh * 9 + kw) * 64 + kc;            // kc mult of 8 -> 16B aligned
}

__global__ __launch_bounds__(256) void k_conv3(const __bf16* __restrict__ in,
                                               const __bf16* __restrict__ wT, // [64][576]
                                               const float* __restrict__ bias,
                                               __bf16* __restrict__ out) {
  __shared__ __bf16 wt[64 * 288];                  // [n][k_local], 36 KB
  const int lane = threadIdx.x & 31;
  const int pair = blockIdx.x * 8 + (threadIdx.x >> 5);
  const int mt0 = pair * 2, mt1 = mt0 + 1;
  const int row = lane & 15, khalf = (lane >> 4) * 8;
  const int col = lane & 15, kb = (lane >> 4) * 8;

  int mA = mt0 * 16 + row, mB = mt1 * 16 + row;
  int btA = mA / 49, pA = mA - btA * 49, oyA = pA / 7, oxA = pA - oyA * 7;
  int btB = mB / 49, pB = mB - btB * 49, oyB = pB / 7, oxB = pB - oyB * 7;
  const __bf16* srcA = in + (((btA * 9 + oyA) * 9) + oxA) * 64;
  const __bf16* srcB = in + (((btB * 9 + oyB) * 9) + oxB) * 64;

  v8f acc[2][4] = {{{}, {}, {}, {}}, {{}, {}, {}, {}}};
  for (int half = 0; half < 2; ++half) {
    __syncthreads();                               // previous-half reads done
    if (threadIdx.x < 32) {                        // 2D tile: 288x64 of 576-wide
      tdm_load_2d_bf16(lds_addr_of(wt), wT + half * 288, 288, 64, 576, 64, 576);
      __builtin_amdgcn_s_wait_tensorcnt((short)0);
    }
    __syncthreads();
    for (int ks = 0; ks < 9; ++ks) {
      int k0 = half * 288 + ks * 32 + khalf;
      v16bf fa0 = ld_frag2(c3addr(srcA, k0), c3addr(srcA, k0 + 16));
      v16bf fa1 = ld_frag2(c3addr(srcB, k0), c3addr(srcB, k0 + 16));
#pragma unroll
      for (int f = 0; f < 4; ++f) {
        const __bf16* brow = wt + (f * 16 + col) * 288 + ks * 32 + kb;
        v16bf fb = ld_frag2(brow, brow + 16);
        acc[0][f] = wmma_bf16(fa0, fb, acc[0][f]);
        acc[1][f] = wmma_bf16(fa1, fb, acc[1][f]);
      }
    }
  }
  const int rb = (lane >> 4) * 8;
#pragma unroll
  for (int f = 0; f < 4; ++f) {
    float bz = bias[f * 16 + col];
#pragma unroll
    for (int v = 0; v < 8; ++v) {                  // x3 layout == [bt][3136]
      float z0 = acc[0][f][v] + bz, z1 = acc[1][f][v] + bz;
      st_bf(out + (mt0 * 16 + rb + v) * 64 + f * 16 + col, z0 > 0.f ? z0 : 0.f);
      st_bf(out + (mt1 * 16 + rb + v) * 64 + f * 16 + col, z1 > 0.f ? z1 : 0.f);
    }
  }
}

// ---------------------------------------------------------------------------
// 4) dense: [2048,3136] x [3136,512] + b, relu. 2 M-tiles x 64 cols per wave.
// ---------------------------------------------------------------------------
__global__ __launch_bounds__(256) void k_dense(const __bf16* __restrict__ x,
                                               const __bf16* __restrict__ wT,  // [512][3136]
                                               const float* __restrict__ bias,
                                               float* __restrict__ outF,
                                               __bf16* __restrict__ featsb) {
  const int lane = threadIdx.x & 31;
  const int n0 = (threadIdx.x >> 5) * 64;
  const int mt0 = blockIdx.x * 2, mt1 = mt0 + 1;
  const int row = lane & 15, khalf = (lane >> 4) * 8;
  const int col = lane & 15, kb = (lane >> 4) * 8;
  const __bf16* arowA = x + (mt0 * 16 + row) * 3136;
  const __bf16* arowB = x + (mt1 * 16 + row) * 3136;

  v8f acc[2][4] = {{{}, {}, {}, {}}, {{}, {}, {}, {}}};
  for (int ks = 0; ks < 98; ++ks) {
    int k0 = ks * 32 + khalf;
    v16bf fa0 = ld_frag2(arowA + k0, arowA + k0 + 16);
    v16bf fa1 = ld_frag2(arowB + k0, arowB + k0 + 16);
    if (ks + 16 < 98) {                            // uniform cond; next chunks
      __builtin_prefetch(arowA + k0 + 512, 0, 0);
      __builtin_prefetch(arowB + k0 + 512, 0, 0);
    }
#pragma unroll
    for (int f = 0; f < 4; ++f) {
      const __bf16* brow = wT + (size_t)(n0 + f * 16 + col) * 3136 + ks * 32 + kb;
      v16bf fb = ld_frag2(brow, brow + 16);
      acc[0][f] = wmma_bf16(fa0, fb, acc[0][f]);
      acc[1][f] = wmma_bf16(fa1, fb, acc[1][f]);
    }
  }
  const int rb = (lane >> 4) * 8;
#pragma unroll
  for (int f = 0; f < 4; ++f) {
    int n = n0 + f * 16 + col;
    float bz = bias[n];
#pragma unroll
    for (int v = 0; v < 8; ++v) {
      int r0 = mt0 * 16 + rb + v, r1 = mt1 * 16 + rb + v;
      float z0 = acc[0][f][v] + bz; z0 = z0 > 0.f ? z0 : 0.f;
      float z1 = acc[1][f][v] + bz; z1 = z1 > 0.f ? z1 : 0.f;
      outF[(size_t)r0 * 576 + n] = z0;             // feats part of output
      outF[(size_t)r1 * 576 + n] = z1;
      st_bf(featsb + r0 * 512 + n, z0);
      st_bf(featsb + r1 * 512 + n, z1);
    }
  }
}

// ---------------------------------------------------------------------------
// 5) mx = feats @ gru_k + bi : [2048,512]x[512,192]. 2 M-tiles, 48 cols/wave.
// ---------------------------------------------------------------------------
__global__ __launch_bounds__(128) void k_mx(const __bf16* __restrict__ feats,
                                            const __bf16* __restrict__ gkT,   // [192][512]
                                            const float* __restrict__ gru_b,
                                            float* __restrict__ mx) {
  const int lane = threadIdx.x & 31;
  const int n0 = (threadIdx.x >> 5) * 48;
  const int mt0 = blockIdx.x * 2, mt1 = mt0 + 1;
  const int row = lane & 15, khalf = (lane >> 4) * 8;
  const int col = lane & 15, kb = (lane >> 4) * 8;
  const __bf16* arowA = feats + (mt0 * 16 + row) * 512;
  const __bf16* arowB = feats + (mt1 * 16 + row) * 512;

  v8f acc[2][3] = {{{}, {}, {}}, {{}, {}, {}}};
  for (int ks = 0; ks < 16; ++ks) {
    int k0 = ks * 32 + khalf;
    v16bf fa0 = ld_frag2(arowA + k0, arowA + k0 + 16);
    v16bf fa1 = ld_frag2(arowB + k0, arowB + k0 + 16);
#pragma unroll
    for (int f = 0; f < 3; ++f) {
      const __bf16* brow = gkT + (n0 + f * 16 + col) * 512 + ks * 32 + kb;
      v16bf fb = ld_frag2(brow, brow + 16);
      acc[0][f] = wmma_bf16(fa0, fb, acc[0][f]);
      acc[1][f] = wmma_bf16(fa1, fb, acc[1][f]);
    }
  }
  const int rb = (lane >> 4) * 8;
#pragma unroll
  for (int f = 0; f < 3; ++f) {
    int n = n0 + f * 16 + col;
    float bz = gru_b[n];                           // bi (row 0)
#pragma unroll
    for (int v = 0; v < 8; ++v) {
      mx[(mt0 * 16 + rb + v) * 192 + n] = acc[0][f][v] + bz;
      mx[(mt1 * 16 + rb + v) * 192 + n] = acc[1][f][v] + bz;
    }
  }
}

// ---------------------------------------------------------------------------
// 6) GRU recurrence: one block, 512 threads (12 WMMA waves + gate math).
// ---------------------------------------------------------------------------
__global__ __launch_bounds__(512) void k_gru(const float* __restrict__ mx,     // [2048][192]
                                             const __bf16* __restrict__ rkT,   // [192][64]
                                             const float* __restrict__ gru_b,
                                             const int* __restrict__ dones,    // [16][128]
                                             const float* __restrict__ state0, // [16][64]
                                             const int* __restrict__ step0,    // [16]
                                             float* __restrict__ dout) {
  __shared__ float hbuf[16 * 64];
  __shared__ float mi[16 * 192];
  __shared__ int stepb[16];
  __shared__ int resetf[16];
  const int tid = threadIdx.x, lane = tid & 31, wave = tid >> 5;
  for (int i = tid; i < 1024; i += 512) hbuf[i] = state0[i];
  if (tid < 16) stepb[tid] = step0[tid];
  __syncthreads();
  const float* br = gru_b + 192;

  for (int t = 0; t < 128; ++t) {
    if (wave < 12) {                               // wave-uniform branch
      const int col = lane & 15, kb = (lane >> 4) * 8;
      const int row = lane & 15, khalf = (lane >> 4) * 8;
      float bz = br[wave * 16 + col];
      v8f acc = {bz, bz, bz, bz, bz, bz, bz, bz};
      const __bf16* rkrow = rkT + (wave * 16 + col) * 64;
#pragma unroll
      for (int kc2 = 0; kc2 < 2; ++kc2) {
        FragBits fa;
#pragma unroll
        for (int g = 0; g < 2; ++g) {              // two 8-float runs -> cvt_pk
          const float* p = hbuf + row * 64 + kc2 * 32 + khalf + g * 16;
          float4 t0 = *(const float4*)p;
          float4 t1 = *(const float4*)(p + 4);
          fa.u[g * 4 + 0] = cvtpk(t0.x, t0.y);
          fa.u[g * 4 + 1] = cvtpk(t0.z, t0.w);
          fa.u[g * 4 + 2] = cvtpk(t1.x, t1.y);
          fa.u[g * 4 + 3] = cvtpk(t1.z, t1.w);
        }
        v16bf fb = ld_frag2(rkrow + kc2 * 32 + kb, rkrow + kc2 * 32 + kb + 16);
        acc = wmma_bf16(fa.b, fb, acc);
      }
      const int rb = (lane >> 4) * 8;
#pragma unroll
      for (int v = 0; v < 8; ++v)
        mi[(rb + v) * 192 + wave * 16 + col] = acc[v];
    }
    __syncthreads();

    float hn[2];
#pragma unroll
    for (int it = 0; it < 2; ++it) {
      int i = tid + it * 512;                      // 0..1023 over (b,u)
      int b = i >> 6, u = i & 63;
      int bt = b * 128 + t;
      const float* mrow = mx + (size_t)bt * 192;
      float z  = 1.f / (1.f + __expf(-(mrow[u]      + mi[b * 192 + u])));
      float r  = 1.f / (1.f + __expf(-(mrow[64 + u] + mi[b * 192 + 64 + u])));
      float hh = tanhf(mrow[128 + u] + r * mi[b * 192 + 128 + u]);
      float hv = z * hbuf[i] + (1.f - z) * hh;     // pre-reset output
      hn[it] = hv;
      dout[(size_t)bt * 576 + 512 + u] = hv;
    }
    __syncthreads();
    if (tid < 16) {
      int s = stepb[tid] + 1;
      int rs = (dones[tid * 128 + t] == 1) || ((s & 15) == 0);
      stepb[tid] = rs ? 0 : s;
      resetf[tid] = rs;
    }
    __syncthreads();
#pragma unroll
    for (int it = 0; it < 2; ++it) {
      int i = tid + it * 512;
      hbuf[i] = resetf[i >> 6] ? 0.f : hn[it];
    }
    __syncthreads();
  }
  const size_t so = (size_t)16 * 128 * 576;
  for (int i = tid; i < 1024; i += 512) dout[so + i] = hbuf[i];
  if (tid < 16) dout[so + 1024 + tid] = (float)stepb[tid];
}

// ---------------------------------------------------------------------------
// launcher
// ---------------------------------------------------------------------------
extern "C" void kernel_launch(void* const* d_in, const int* in_sizes, int n_in,
                              void* d_out, int out_size, void* d_ws, size_t ws_size,
                              hipStream_t stream) {
  (void)in_sizes; (void)n_in; (void)out_size; (void)ws_size;
  const float* inpt   = (const float*)d_in[0];
  const int*   dones  = (const int*)d_in[1];
  const float* state0 = (const float*)d_in[2];
  const int*   step0  = (const int*)d_in[3];
  const float* c1w = (const float*)d_in[4];
  const float* c1b = (const float*)d_in[5];
  const float* c2w = (const float*)d_in[6];
  const float* c2b = (const float*)d_in[7];
  const float* c3w = (const float*)d_in[8];
  const float* c3b = (const float*)d_in[9];
  const float* dw  = (const float*)d_in[10];
  const float* db  = (const float*)d_in[11];
  const float* gk  = (const float*)d_in[12];
  const float* grk = (const float*)d_in[13];
  const float* gb  = (const float*)d_in[14];

  char* ws = (char*)d_ws;                          // ~93.8 MB total
  __bf16* x1   = (__bf16*)(ws + 0);                // [2048][20][20][32]
  __bf16* x2   = (__bf16*)(ws + 52428800);         // [2048][9][9][64]
  __bf16* x3   = (__bf16*)(ws + 73662464);         // [2048][3136]
  __bf16* fb   = (__bf16*)(ws + 86507520);         // feats bf16 [2048][512]
  float*  mx   = (float*) (ws + 88604672);         // [2048][192]
  __bf16* dwT  = (__bf16*)(ws + 90177536);         // dense_w^T bf16 [512][3136]
  __bf16* gkT  = (__bf16*)(ws + 93388800);         // gru_k^T  bf16 [192][512]
  __bf16* grkT = (__bf16*)(ws + 93585408);         // gru_rk^T bf16 [192][64]
  __bf16* c1T  = (__bf16*)(ws + 93609984);         // conv1 w^T bf16 [32][256]
  __bf16* c2T  = (__bf16*)(ws + 93626368);         // conv2 w^T bf16 [64][512]
  __bf16* c3T  = (__bf16*)(ws + 93691904);         // conv3 w^T bf16 [64][576]
  float* out = (float*)d_out;

  k_cvtT<<<512, 256, 0, stream>>>(dw, dwT, 3136, 512);
  k_cvtT<<<64, 256, 0, stream>>>(gk, gkT, 512, 192);
  k_cvtT<<<8, 256, 0, stream>>>(grk, grkT, 64, 192);
  k_cvtT<<<8, 256, 0, stream>>>(c1w, c1T, 256, 32);
  k_cvtT<<<16, 256, 0, stream>>>(c2w, c2T, 512, 64);
  k_cvtT<<<16, 256, 0, stream>>>(c3w, c3T, 576, 64);

  k_conv1<<<3200, 256, 0, stream>>>(inpt, c1T, c1b, x1);    // 25600 tile-pairs
  k_conv2<<<648, 256, 0, stream>>>(x1, c2T, c2b, x2);       // 5184 tile-pairs
  k_conv3<<<392, 256, 0, stream>>>(x2, c3T, c3b, x3);       // 3136 tile-pairs
  k_dense<<<64, 256, 0, stream>>>(x3, dwT, db, out, fb);
  k_mx<<<64, 128, 0, stream>>>(fb, gkT, gb, mx);
  k_gru<<<1, 512, 0, stream>>>(mx, grkT, gb, dones, state0, step0, out);
}
